// SwinTransformerBlock_12713103197236
// MI455X (gfx1250) — matrix operand order repaired and checked by reading, compiled, not run on hardware
//
#include <hip/hip_runtime.h>

typedef __attribute__((ext_vector_type(16))) _Float16 v16h;
typedef __attribute__((ext_vector_type(8)))  _Float16 v8h;
typedef __attribute__((ext_vector_type(8)))  float    v8f;

// Problem constants (Swin block, B=16, 56x56, C=384, 12 heads, 7x7 windows, shift 3)
static constexpr int C_    = 384;
static constexpr int NH    = 12;
static constexpr int HDIM  = 32;
static constexpr int HIMG  = 56;
static constexpr int BIMG  = 16;
static constexpr int LTOK  = BIMG * HIMG * HIMG;   // 50176
static constexpr int NWIN  = BIMG * 8 * 8;         // 1024 windows
static constexpr int NPAD  = 64;                   // 49 tokens padded to 64
static constexpr int MWIN  = NWIN * NPAD;          // 65536 window rows
static constexpr int HID   = 1536;

// ---------------- fragment helpers (ISA 7.12.2 layouts, wave32) ----------------
// A (16x32 f16): lane half -> K octet selection
__device__ __forceinline__ v16h load_a(const _Float16* row_ptr, int half) {
  v8h c0 = *(const v8h*)(row_ptr + half * 8);
  v8h c1 = *(const v8h*)(row_ptr + 16 + half * 8);
  v16h r;
#pragma unroll
  for (int i = 0; i < 8; i++) { r[i] = c0[i]; r[8 + i] = c1[i]; }
  return r;
}
// B (32x16 f16) from a transposed (N-major, K-contiguous) row: lane half -> K octet selection
__device__ __forceinline__ v16h load_b(const _Float16* bt_row, int half) {
  v8h c0 = *(const v8h*)(bt_row + half * 8);
  v8h c1 = *(const v8h*)(bt_row + 16 + half * 8);
  v16h r;
#pragma unroll
  for (int i = 0; i < 8; i++) { r[i] = c0[i]; r[8 + i] = c1[i]; }
  return r;
}

// ---------------- weight convert + transpose: W(K,N) f32 -> Wt(N,K) f16 --------
__global__ void wcvt_kernel(const float* __restrict__ src, _Float16* __restrict__ dst,
                            int K, int N) {
  int t = blockIdx.x * blockDim.x + threadIdx.x;
  if (t >= K * N) return;
  int n = t % N, k = t / N;
  dst[(size_t)n * K + k] = (_Float16)src[t];
}

// ---------------- LN1 fused with shifted-window partition ----------------------
// One wave (32 lanes) per padded window row; rows >=49 are zeroed.
__global__ __launch_bounds__(256) void ln1_win_kernel(
    const float* __restrict__ x, const float* __restrict__ g,
    const float* __restrict__ b, _Float16* __restrict__ hwin) {
  int wave = threadIdx.x >> 5, lane = threadIdx.x & 31;
  int gidx = blockIdx.x * 8 + wave;            // 0..MWIN-1
  int win = gidx >> 6, r = gidx & 63;
  _Float16* dst = hwin + (size_t)gidx * C_;
  if (r >= 49) {
    for (int j = lane; j < C_; j += 32) dst[j] = (_Float16)0.f;
    return;
  }
  int bi = win >> 6, wy = (win >> 3) & 7, wx = win & 7;
  int ih = r / 7, iw = r % 7;
  int gh = (wy * 7 + ih + 3) % HIMG;           // inverse of roll(-3,-3)
  int gw = (wx * 7 + iw + 3) % HIMG;
  const float* src = x + ((size_t)bi * (HIMG * HIMG) + gh * HIMG + gw) * C_;
  float v[12], s = 0.f;
#pragma unroll
  for (int t = 0; t < 12; t++) { v[t] = src[lane + 32 * t]; s += v[t]; }
  for (int o = 16; o >= 1; o >>= 1) s += __shfl_xor(s, o, 32);
  float mu = s * (1.f / C_), vs = 0.f;
#pragma unroll
  for (int t = 0; t < 12; t++) { float d = v[t] - mu; vs += d * d; }
  for (int o = 16; o >= 1; o >>= 1) vs += __shfl_xor(vs, o, 32);
  float rs = rsqrtf(vs * (1.f / C_) + 1e-5f);
#pragma unroll
  for (int t = 0; t < 12; t++) {
    int j = lane + 32 * t;
    dst[j] = (_Float16)((v[t] - mu) * rs * g[j] + b[j]);
  }
}

// ---------------- LN2 over residual x1 (in d_out) -> f16 -----------------------
__global__ __launch_bounds__(256) void ln2_kernel(
    const float* __restrict__ xr, const float* __restrict__ g,
    const float* __restrict__ b, _Float16* __restrict__ h2) {
  int wave = threadIdx.x >> 5, lane = threadIdx.x & 31;
  int row = blockIdx.x * 8 + wave;
  const float* src = xr + (size_t)row * C_;
  _Float16* dst = h2 + (size_t)row * C_;
  float v[12], s = 0.f;
#pragma unroll
  for (int t = 0; t < 12; t++) { v[t] = src[lane + 32 * t]; s += v[t]; }
  for (int o = 16; o >= 1; o >>= 1) s += __shfl_xor(s, o, 32);
  float mu = s * (1.f / C_), vs = 0.f;
#pragma unroll
  for (int t = 0; t < 12; t++) { float d = v[t] - mu; vs += d * d; }
  for (int o = 16; o >= 1; o >>= 1) vs += __shfl_xor(vs, o, 32);
  float rs = rsqrtf(vs * (1.f / C_) + 1e-5f);
#pragma unroll
  for (int t = 0; t < 12; t++) {
    int j = lane + 32 * t;
    dst[j] = (_Float16)((v[t] - mu) * rs * g[j] + b[j]);
  }
}

// ---------------- 128x128-tile WMMA GEMM, double-buffered LDS pipeline ---------
// C[M,N] = A[M,K](f16) * Wt[N,K]^T (f16), f32 accumulate. 256 thr = 8 waves.
// Wave w owns a 64x32 sub-tile: rows (w>>2)*64 + [0,64), cols (w&3)*32 + [0,32)
// -> 8 WMMAs per 32-wide K-step. Next K-tile's global_load_b128 x4 are issued
// into 4 distinct register temps one iteration ahead (clause-batchable),
// overlapped with the WMMA burst; one barrier per K-step.
// epi: 0 = +bias -> f16 ; 1 = +bias, exact GELU -> f16 ;
//      2 = +bias, reverse-shift window scatter + residual(x) -> f32 d_out ;
//      3 = +bias, += into f32 d_out (residual x1 already there).
__global__ __launch_bounds__(256) void gemm_kernel(
    const _Float16* __restrict__ A, const _Float16* __restrict__ BT,
    const float* __restrict__ bias, int M, int N, int K, int epi,
    _Float16* __restrict__ outh, float* __restrict__ outf,
    const float* __restrict__ xres) {
  __shared__ __align__(16) _Float16 As[2][128 * 32];
  __shared__ __align__(16) _Float16 Bs[2][128 * 32];
  int wv = threadIdx.x >> 5, lane = threadIdx.x & 31;
  int half = lane >> 4, l16 = lane & 15;
  int m0 = blockIdx.y * 128, n0 = blockIdx.x * 128;
  int wm = (wv >> 2) * 64;      // wave row offset inside tile
  int wn = (wv & 3) * 32;       // wave col offset inside tile
  v8f acc[4][2];
#pragma unroll
  for (int i = 0; i < 4; i++)
#pragma unroll
    for (int n = 0; n < 2; n++)
#pragma unroll
      for (int j = 0; j < 8; j++) acc[i][n][j] = 0.f;

  // per-thread staging addresses (tile = 4096 f16, 8 f16 x 2 chunks per thread)
  int idx  = threadIdx.x * 8;
  int row  = idx >> 5,  kk  = idx & 31;
  int idx2 = idx + 2048;
  int row2 = idx2 >> 5, kk2 = idx2 & 31;
  const _Float16* Ap  = A  + (size_t)(m0 + row)  * K + kk;
  const _Float16* Ap2 = A  + (size_t)(m0 + row2) * K + kk2;
  const _Float16* Bp  = BT + (size_t)(n0 + row)  * K + kk;
  const _Float16* Bp2 = BT + (size_t)(n0 + row2) * K + kk2;

  // prologue: stage K-tile 0 into buffer 0
  {
    v8h ta0 = *(const v8h*)Ap;
    v8h ta1 = *(const v8h*)Ap2;
    v8h tb0 = *(const v8h*)Bp;
    v8h tb1 = *(const v8h*)Bp2;
    *(v8h*)(As[0] + idx)  = ta0;
    *(v8h*)(As[0] + idx2) = ta1;
    *(v8h*)(Bs[0] + idx)  = tb0;
    *(v8h*)(Bs[0] + idx2) = tb1;
  }

  int nsteps = K / 32;
  for (int s = 0; s < nsteps; s++) {
    int buf = s & 1;
    v8h ta0, ta1, tb0, tb1;
    bool more = (s + 1 < nsteps);
    if (more) {  // issue next tile's global loads into 4 independent temps
      int ko = (s + 1) * 32;
      ta0 = *(const v8h*)(Ap + ko);
      ta1 = *(const v8h*)(Ap2 + ko);
      tb0 = *(const v8h*)(Bp + ko);
      tb1 = *(const v8h*)(Bp2 + ko);
      if (s + 2 < nsteps) {  // gfx1250 global_prefetch_b8, two tiles ahead
        __builtin_prefetch(Ap + (s + 2) * 32, 0, 1);
        __builtin_prefetch(Bp + (s + 2) * 32, 0, 1);
      }
    }
    __syncthreads();  // buf stores (prev iter) visible; buf^1 readers done
    if (more) {
      _Float16* an = As[buf ^ 1];
      _Float16* bn = Bs[buf ^ 1];
      *(v8h*)(an + idx)  = ta0;
      *(v8h*)(an + idx2) = ta1;
      *(v8h*)(bn + idx)  = tb0;
      *(v8h*)(bn + idx2) = tb1;
    }
    // load ALL fragments, then 8 back-to-back WMMAs (D->C accumulate chains
    // need no hazard NOPs).
    const _Float16* as = As[buf];
    const _Float16* bs = Bs[buf];
    v16h bf0 = load_b(bs + (wn + l16) * 32, half);
    v16h bf1 = load_b(bs + (wn + 16 + l16) * 32, half);
    v16h af[4];
#pragma unroll
    for (int mt = 0; mt < 4; mt++)
      af[mt] = load_a(as + (wm + mt * 16 + l16) * 32, half);
#pragma unroll
    for (int mt = 0; mt < 4; mt++) {
      acc[mt][0] = __builtin_amdgcn_wmma_f32_16x16x32_f16(
          false, af[mt], false, bf0, (short)0, acc[mt][0], false, false);
      acc[mt][1] = __builtin_amdgcn_wmma_f32_16x16x32_f16(
          false, af[mt], false, bf1, (short)0, acc[mt][1], false, false);
    }
  }

#pragma unroll
  for (int nt = 0; nt < 2; nt++) {
    int col = n0 + wn + nt * 16 + l16;
    float bv = bias[col];
#pragma unroll
    for (int mt = 0; mt < 4; mt++) {
#pragma unroll
      for (int r = 0; r < 8; r++) {
        int orow = m0 + wm + mt * 16 + half * 8 + r;
        float val = acc[mt][nt][r] + bv;
        if (epi == 0) {
          outh[(size_t)orow * N + col] = (_Float16)val;
        } else if (epi == 1) {
          val = 0.5f * val * (1.f + erff(val * 0.70710678118654752f));
          outh[(size_t)orow * N + col] = (_Float16)val;
        } else if (epi == 2) {
          int win = orow >> 6, rl = orow & 63;
          if (rl < 49) {
            int bi = win >> 6, wy = (win >> 3) & 7, wx = win & 7;
            int ih = rl / 7, iw = rl % 7;
            int gh = (wy * 7 + ih + 3) % HIMG, gw = (wx * 7 + iw + 3) % HIMG;
            size_t oidx = ((size_t)bi * (HIMG * HIMG) + gh * HIMG + gw) * C_ + col;
            outf[oidx] = xres[oidx] + val;
          }
        } else {
          size_t oidx = (size_t)orow * N + col;
          outf[oidx] += val;
        }
      }
    }
  }
}

// ---------------- windowed attention: one block per (window, head) -------------
__global__ __launch_bounds__(128) void attn_kernel(
    const _Float16* __restrict__ qkv, const float* __restrict__ rpb,
    _Float16* __restrict__ out) {
  __shared__ __align__(16) _Float16 Kt[64 * 32];   // K rows x d
  __shared__ __align__(16) _Float16 Vt[32 * 64];   // V transposed: d x rows
  __shared__ __align__(16) float    Sm[64 * 64];
  __shared__ __align__(16) _Float16 Pm[64 * 64];
  int wv = threadIdx.x >> 5, lane = threadIdx.x & 31;
  int half = lane >> 4, l16 = lane & 15;
  int win = blockIdx.x / NH, head = blockIdx.x % NH;
  const _Float16* base = qkv + (size_t)win * NPAD * (3 * C_) + head * HDIM;

  for (int idx = threadIdx.x; idx < 64 * 32; idx += 128) {
    int row = idx >> 5, d = idx & 31;
    Kt[idx]          = base[(size_t)row * (3 * C_) + C_ + d];
    Vt[d * 64 + row] = base[(size_t)row * (3 * C_) + 2 * C_ + d];
  }
  __syncthreads();

  // q fragment (this wave's 16 rows), pre-scaled by 1/sqrt(hd)
  const float scale = 0.17677669529663687f;
  v16h qa;
  {
    const _Float16* qp = base + (size_t)(wv * 16 + l16) * (3 * C_);
    v8h c0 = *(const v8h*)(qp + half * 8);
    v8h c1 = *(const v8h*)(qp + 16 + half * 8);
#pragma unroll
    for (int i = 0; i < 8; i++) {
      qa[i]     = (_Float16)((float)c0[i] * scale);
      qa[8 + i] = (_Float16)((float)c1[i] * scale);
    }
  }
  int wy = (win >> 3) & 7, wx = win & 7;

  // S = q*k^T + rel-pos bias + shift mask (bias/mask injected via C fragment)
  for (int nt = 0; nt < 4; nt++) {
    v16h kb = load_b(Kt + (nt * 16 + l16) * 32, half);
    v8f cf;
    int cn = nt * 16 + l16;
#pragma unroll
    for (int r = 0; r < 8; r++) {
      int mr = wv * 16 + half * 8 + r;
      float cv;
      if (mr >= 49 || cn >= 49) {
        cv = -1e9f;
      } else {
        int ih = mr / 7, iw = mr % 7, jh = cn / 7, jw = cn % 7;
        cv = rpb[((ih - jh + 6) * 13 + (iw - jw + 6)) * NH + head];
        int ghi = wy * 7 + ih, gwi = wx * 7 + iw;
        int ghj = wy * 7 + jh, gwj = wx * 7 + jw;
        int ri = (ghi < 49 ? 0 : (ghi < 53 ? 1 : 2)) * 3 + (gwi < 49 ? 0 : (gwi < 53 ? 1 : 2));
        int rj = (ghj < 49 ? 0 : (ghj < 53 ? 1 : 2)) * 3 + (gwj < 49 ? 0 : (gwj < 53 ? 1 : 2));
        if (ri != rj) cv -= 100.f;
      }
      cf[r] = cv;
    }
    cf = __builtin_amdgcn_wmma_f32_16x16x32_f16(false, qa, false, kb, (short)0, cf,
                                                false, false);
#pragma unroll
    for (int r = 0; r < 8; r++) Sm[(wv * 16 + half * 8 + r) * 64 + cn] = cf[r];
  }
  __builtin_amdgcn_wave_barrier();  // Sm rows are wave-private; order LDS ops

  // in-wave softmax: lane pair (l16, l16+16) covers cols [0,32) / [32,49)
  {
    int row = wv * 16 + l16;
    int c0 = half * 32, cend = half ? 49 : 32;
    float mx = -1e30f;
    for (int c = c0; c < cend; c++) mx = fmaxf(mx, Sm[row * 64 + c]);
    mx = fmaxf(mx, __shfl_xor(mx, 16, 32));
    float sum = 0.f;
    for (int c = c0; c < cend; c++) {
      float e = __expf(Sm[row * 64 + c] - mx);
      sum += e;
      Sm[row * 64 + c] = e;
    }
    sum += __shfl_xor(sum, 16, 32);
    float inv = 1.f / sum;
    for (int c = c0; c < c0 + 32; c++)
      Pm[row * 64 + c] = (_Float16)((c < 49) ? Sm[row * 64 + c] * inv : 0.f);
  }
  __builtin_amdgcn_wave_barrier();

  // O = P(16x64) @ V(64x32): two k=32 WMMA steps, two 16-col tiles
  for (int nt = 0; nt < 2; nt++) {
    v8f of;
#pragma unroll
    for (int r = 0; r < 8; r++) of[r] = 0.f;
#pragma unroll
    for (int ks = 0; ks < 2; ks++) {
      v16h pa = load_a(Pm + (wv * 16 + l16) * 64 + ks * 32, half);
      v16h vb = load_b(Vt + (nt * 16 + l16) * 64 + ks * 32, half);
      of = __builtin_amdgcn_wmma_f32_16x16x32_f16(false, pa, false, vb, (short)0, of,
                                                  false, false);
    }
#pragma unroll
    for (int r = 0; r < 8; r++) {
      int mr = wv * 16 + half * 8 + r;
      out[((size_t)win * NPAD + mr) * C_ + head * HDIM + nt * 16 + l16] =
          (_Float16)of[r];
    }
  }
}

// --------------------------------- launch --------------------------------------
extern "C" void kernel_launch(void* const* d_in, const int* in_sizes, int n_in,
                              void* d_out, int out_size, void* d_ws, size_t ws_size,
                              hipStream_t stream) {
  const float* x     = (const float*)d_in[0];
  const float* n1g   = (const float*)d_in[3];
  const float* n1b   = (const float*)d_in[4];
  const float* qkvw  = (const float*)d_in[5];
  const float* qkvb  = (const float*)d_in[6];
  const float* rpb   = (const float*)d_in[7];
  const float* projw = (const float*)d_in[8];
  const float* projb = (const float*)d_in[9];
  const float* n2g   = (const float*)d_in[10];
  const float* n2b   = (const float*)d_in[11];
  const float* w1    = (const float*)d_in[12];
  const float* b1    = (const float*)d_in[13];
  const float* w2    = (const float*)d_in[14];
  const float* b2    = (const float*)d_in[15];
  float* out = (float*)d_out;

  char* ws = (char*)d_ws;
  size_t off = 0;
  auto carve = [&](size_t bytes) {
    void* p = ws + off;
    off = (off + bytes + 255) & ~(size_t)255;
    return p;
  };
  _Float16* WqkvT = (_Float16*)carve((size_t)(3 * C_) * C_ * 2);  // 1152x384
  _Float16* WprT  = (_Float16*)carve((size_t)C_ * C_ * 2);        // 384x384
  _Float16* W1T   = (_Float16*)carve((size_t)HID * C_ * 2);       // 1536x384
  _Float16* W2T   = (_Float16*)carve((size_t)C_ * HID * 2);       // 384x1536
  _Float16* bufA  = (_Float16*)carve((size_t)MWIN * C_ * 2);      // hwin / attn-out / h2
  _Float16* bufB  = (_Float16*)carve((size_t)LTOK * HID * 2);     // qkv / mlp hidden

  // 1) weight convert + transpose (f32 -> f16, N-major)
  wcvt_kernel<<<(C_ * 3 * C_ + 255) / 256, 256, 0, stream>>>(qkvw, WqkvT, C_, 3 * C_);
  wcvt_kernel<<<(C_ * C_ + 255) / 256, 256, 0, stream>>>(projw, WprT, C_, C_);
  wcvt_kernel<<<(C_ * HID + 255) / 256, 256, 0, stream>>>(w1, W1T, C_, HID);
  wcvt_kernel<<<(HID * C_ + 255) / 256, 256, 0, stream>>>(w2, W2T, HID, C_);

  // 2) LN1 + shifted-window partition -> bufA (f16, MWIN x 384)
  ln1_win_kernel<<<MWIN / 8, 256, 0, stream>>>(x, n1g, n1b, bufA);

  // 3) QKV GEMM: (MWIN x 384) * (384 x 1152) -> bufB f16
  gemm_kernel<<<dim3(3 * C_ / 128, MWIN / 128), 256, 0, stream>>>(
      bufA, WqkvT, qkvb, MWIN, 3 * C_, C_, 0, bufB, nullptr, nullptr);

  // 4) windowed attention -> bufA f16 (MWIN x 384)
  attn_kernel<<<NWIN * NH, 128, 0, stream>>>(bufB, rpb, bufA);

  // 5) proj GEMM + reverse shift scatter + residual -> d_out (x1, f32)
  gemm_kernel<<<dim3(C_ / 128, MWIN / 128), 256, 0, stream>>>(
      bufA, WprT, projb, MWIN, C_, C_, 2, nullptr, out, x);

  // 6) LN2 -> bufA f16 (LTOK x 384)
  ln2_kernel<<<LTOK / 8, 256, 0, stream>>>(out, n2g, n2b, bufA);

  // 7) MLP1 + GELU -> bufB f16 (LTOK x 1536)
  gemm_kernel<<<dim3(HID / 128, LTOK / 128), 256, 0, stream>>>(
      bufA, W1T, b1, LTOK, HID, C_, 1, bufB, nullptr, nullptr);

  // 8) MLP2 + residual add into d_out
  gemm_kernel<<<dim3(C_ / 128, LTOK / 128), 256, 0, stream>>>(
      bufB, W2T, b2, LTOK, C_, HID, 3, nullptr, out, nullptr);
}
